// FANMoE_HyperNetLinearLayer_6399501271820
// MI455X (gfx1250) — matrix operand order repaired and checked by reading, not compiled, not run on hardware
//
#include <hip/hip_runtime.h>
#include <hip/hip_bf16.h>

#define B_    512
#define IN_   256
#define OUT_  256
#define COND_ 128
#define NEXP  8
#define HH    64          // hyper hidden (K of h)
#define DP_   64
#define DN_   128
#define TPE_  49280       // IN*DP + IN*DN + DN
#define TOTAL_ 394240     // NEXP * TPE
#define PCOL  192         // DP + DN columns per expert
#define NCOL  (NEXP*PCOL) // 1536
#define KB    (HH*IN_)    // 16384 bf16-GEMM K

typedef __attribute__((ext_vector_type(16))) __bf16 v16bf;
typedef __attribute__((ext_vector_type(8)))  __bf16 v8bf;
typedef __attribute__((ext_vector_type(8)))  float  v8f;
typedef __attribute__((ext_vector_type(2)))  float  v2f;

union AFrag { v16bf v; v8bf h[2]; };

// ---------------------------------------------------------------------------
// Kernel 1: per-batch prep — h = relu(cond@hW1+hb1), gate softmax,
//           Z[b, i*64+k] = x[b,i] * h[b,k]  (bf16)
// ---------------------------------------------------------------------------
__global__ __launch_bounds__(128)
void prep_batch(const float* __restrict__ x, const float* __restrict__ cond,
                const float* __restrict__ hW1, const float* __restrict__ hb1,
                const float* __restrict__ gW1, const float* __restrict__ gb1,
                const float* __restrict__ gW2, const float* __restrict__ gb2,
                __bf16* __restrict__ Z, float* __restrict__ Hout,
                float* __restrict__ GW)
{
    __shared__ float cS[COND_], xS[IN_], hS[HH], gS[3*NEXP], lg[NEXP];
    const int b = blockIdx.x, t = threadIdx.x;
    cS[t]       = cond[b*COND_ + t];
    xS[t]       = x[b*IN_ + t];
    xS[t + 128] = x[b*IN_ + t + 128];
    __syncthreads();
    if (t < HH) {
        float acc = hb1[t];
        for (int j = 0; j < COND_; ++j) acc += cS[j] * hW1[j*HH + t];
        acc = fmaxf(acc, 0.f);
        hS[t] = acc;
        Hout[b*HH + t] = acc;
    }
    if (t < 3*NEXP) {
        float acc = gb1[t];
        for (int j = 0; j < COND_; ++j) acc += cS[j] * gW1[j*(3*NEXP) + t];
        gS[t] = fmaxf(acc, 0.f);
    }
    __syncthreads();
    if (t < NEXP) {
        float acc = gb2[t];
        for (int j = 0; j < 3*NEXP; ++j) acc += gS[j] * gW2[j*NEXP + t];
        lg[t] = acc;
    }
    __syncthreads();
    if (t < NEXP) {
        float m = lg[0];
        for (int j = 1; j < NEXP; ++j) m = fmaxf(m, lg[j]);
        float s = 0.f;
        for (int j = 0; j < NEXP; ++j) s += __expf(lg[j] - m);
        GW[b*NEXP + t] = __expf(lg[t] - m) / s;
    }
    __syncthreads();
    __bf16* Zr = Z + (size_t)b * KB;
    for (int e = t; e < KB; e += 128) {
        const int i = e >> 6, k = e & 63;          // kk = i*64 + k
        Zr[e] = (__bf16)(xS[i] * hS[k]);
    }
}

// ---------------------------------------------------------------------------
// Kernel 2: pack hW2 into K-fastest bf16 B panels:
//           Wb[(n*192+p)*KB + i*64 + k] = hW2[k, n*TPE + colsrc(i,p)]
//           (LDS-tiled transpose, coalesced both sides)
// ---------------------------------------------------------------------------
__global__ __launch_bounds__(256)
void pack_w(const float* __restrict__ hW2, __bf16* __restrict__ Wb)
{
    __shared__ float tile[64 * 197];
    const int i = blockIdx.x;        // 0..255  (input index)
    const int n = blockIdx.y;        // 0..7    (expert)
    const int t = threadIdx.x;
    const size_t nb = (size_t)n * TPE_;
    for (int idx = t; idx < 64 * PCOL; idx += 256) {
        const int k = idx / PCOL, p = idx % PCOL;
        const size_t src = (size_t)k * TOTAL_ + nb +
            (p < DP_ ? (size_t)i * DP_ + p
                     : (size_t)IN_ * DP_ + (size_t)i * DN_ + (p - DP_));
        tile[k * 197 + p] = hW2[src];
    }
    __syncthreads();
    for (int idx = t; idx < 64 * PCOL; idx += 256) {
        const int p = idx >> 6, k = idx & 63;      // k fastest on write
        Wb[(size_t)(n * PCOL + p) * KB + (size_t)i * 64 + k] =
            (__bf16)tile[k * 197 + p];
    }
}

// ---------------------------------------------------------------------------
// Kernel 3: GEMM  C[b,(n,p)] = Z @ Wb  (bf16 WMMA, f32 acc)
//                            + x @ (base + hb2)  (f32 WMMA)
// Block: 128 threads (4 waves), tile 64 rows x 64 cols.
// A staged via CDNA5 async global->LDS copies (ASYNCcnt), no VGPR bounce.
// ---------------------------------------------------------------------------
__global__ __launch_bounds__(128)
void gemm_kernel(const __bf16* __restrict__ Z, const __bf16* __restrict__ Wb,
                 const float* __restrict__ x,
                 const float* __restrict__ base_wp,
                 const float* __restrict__ base_wn,
                 const float* __restrict__ hb2,
                 float* __restrict__ C)
{
    __shared__ __align__(32) __bf16 As[64 * 32];
    const int t = threadIdx.x;
    const int w = t >> 5, lane = t & 31;
    const int lcol = lane & 15, lhalf = lane >> 4;
    const int m0  = blockIdx.x * 64;
    const int col = blockIdx.y * 64 + w * 16 + lcol;   // lane's column

    v8f c[4] = {};

    // per-thread A-stage addresses: thread copies one 32B row-chunk
    const int srow = t >> 1, shalf = t & 1;
    const __bf16* gA0 = Z + (size_t)(m0 + srow) * KB + shalf * 16;
    const unsigned ldsA = (unsigned)(size_t)(void*)(As + srow * 32 + shalf * 16);

    // ---- bf16 hypernet segment (K = 16384), v_wmma_f32_16x16x32_bf16 ----
    for (int kk0 = 0; kk0 < KB; kk0 += 32) {
        __syncthreads();   // previous iteration's fragment reads complete
        {   // async global -> LDS stage: 64 rows x 32 K bf16 (4 KB total)
            const __bf16* gsrc = gA0 + kk0;
            asm volatile(
                "global_load_async_to_lds_b128 %0, %1, off\n\t"
                "global_load_async_to_lds_b128 %0, %1, off offset:16"
                :: "v"(ldsA), "v"(gsrc) : "memory");
        }
        asm volatile("s_wait_asynccnt 0x0" ::: "memory");
        __syncthreads();
        const v16bf bfrag =
            *(const v16bf*)(Wb + (size_t)col * KB + kk0 + lhalf * 16);
#pragma unroll
        for (int r = 0; r < 4; ++r) {
            AFrag a;
            const __bf16* ap = As + (r * 16 + lcol) * 32;
            a.h[0] = *(const v8bf*)(ap + lhalf * 16);       // K 0..7 / 16..23
            a.h[1] = *(const v8bf*)(ap + lhalf * 16 + 8);   // K 8..15 / 24..31
            c[r] = __builtin_amdgcn_wmma_f32_16x16x32_bf16(
                false, a.v, false, bfrag, (short)0, c[r], false, false);
        }
    }

    // ---- f32 base segment (K = 256 over x), v_wmma_f32_16x16x4_f32 ----
    const int n = col / PCOL, p = col % PCOL;
    for (int i0 = 0; i0 < IN_; i0 += 4) {
        const int kin = i0 + lhalf * 2;
        v2f b2;
        if (p < DP_) {
            b2.x = base_wp[(n*IN_ + kin    ) * DP_ + p] +
                   hb2[(size_t)n*TPE_ + (size_t)kin    * DP_ + p];
            b2.y = base_wp[(n*IN_ + kin + 1) * DP_ + p] +
                   hb2[(size_t)n*TPE_ + (size_t)(kin+1) * DP_ + p];
        } else {
            const int q = p - DP_;
            b2.x = base_wn[(n*IN_ + kin    ) * DN_ + q] +
                   hb2[(size_t)n*TPE_ + IN_*DP_ + (size_t)kin    * DN_ + q];
            b2.y = base_wn[(n*IN_ + kin + 1) * DN_ + q] +
                   hb2[(size_t)n*TPE_ + IN_*DP_ + (size_t)(kin+1) * DN_ + q];
        }
#pragma unroll
        for (int r = 0; r < 4; ++r) {
            const int arow = m0 + r * 16 + lcol;
            const v2f a2 = *(const v2f*)(x + (size_t)arow * IN_ + kin);
            c[r] = __builtin_amdgcn_wmma_f32_16x16x4_f32(
                false, a2, false, b2, (short)0, c[r], false, false);
        }
    }

    // ---- store C fragments (f32 C/D layout: m = e + 8*lhalf) ----
#pragma unroll
    for (int r = 0; r < 4; ++r)
#pragma unroll
        for (int e = 0; e < 8; ++e) {
            const int row = m0 + r * 16 + lhalf * 8 + e;
            C[(size_t)row * NCOL + col] = c[r][e];
        }
}

// ---------------------------------------------------------------------------
// Kernel 4: epilogue — bn term (h @ hW2_bn + biases), relu/cos/sin,
//           gate-weighted reduction over the 8 experts.
// ---------------------------------------------------------------------------
__global__ __launch_bounds__(128)
void epilogue(const float* __restrict__ C, const float* __restrict__ Hf,
              const float* __restrict__ GW, const float* __restrict__ hW2,
              const float* __restrict__ base_bn, const float* __restrict__ hb2,
              float* __restrict__ out)
{
    __shared__ float hS[HH], gS[NEXP];
    const int b = blockIdx.x, t = threadIdx.x;
    if (t < HH)   hS[t] = Hf[b*HH + t];
    if (t < NEXP) gS[t] = GW[b*NEXP + t];
    __syncthreads();
    const float* Cr = C + (size_t)b * NCOL;
    if (t < DP_) {          // periodic branch: cos / sin combine
        float ac = 0.f, as = 0.f;
        for (int n = 0; n < NEXP; ++n) {
            const float v = Cr[n * PCOL + t];
            ac += __cosf(v) * gS[n];
            as += __sinf(v) * gS[n];
        }
        out[b*OUT_ + t]       = ac;
        out[b*OUT_ + DP_ + t] = as;
    }
    {                       // non-periodic branch: + bn term, relu, combine
        const int p = t;    // 0..127
        float an = 0.f;
        for (int n = 0; n < NEXP; ++n) {
            const size_t off = (size_t)n*TPE_ + IN_*DP_ + IN_*DN_ + p;
            float v = Cr[n * PCOL + DP_ + p] + base_bn[n*DN_ + p] + hb2[off];
            for (int k = 0; k < HH; ++k)
                v += hS[k] * hW2[(size_t)k * TOTAL_ + off];
            an += fmaxf(v, 0.f) * gS[n];
        }
        out[b*OUT_ + 2*DP_ + p] = an;
    }
}

// ---------------------------------------------------------------------------
extern "C" void kernel_launch(void* const* d_in, const int* in_sizes, int n_in,
                              void* d_out, int out_size, void* d_ws, size_t ws_size,
                              hipStream_t stream)
{
    const float* x       = (const float*)d_in[0];
    const float* cond    = (const float*)d_in[1];
    const float* base_wp = (const float*)d_in[2];
    const float* base_wn = (const float*)d_in[3];
    const float* base_bn = (const float*)d_in[4];
    const float* hW1     = (const float*)d_in[5];
    const float* hb1     = (const float*)d_in[6];
    const float* hW2     = (const float*)d_in[7];
    const float* hb2     = (const float*)d_in[8];
    const float* gW1     = (const float*)d_in[9];
    const float* gb1     = (const float*)d_in[10];
    const float* gW2     = (const float*)d_in[11];
    const float* gb2     = (const float*)d_in[12];
    float* out = (float*)d_out;

    char* ws = (char*)d_ws;
    __bf16* Z  = (__bf16*)ws;                                      // 16 MB
    __bf16* Wb = (__bf16*)(ws + (size_t)B_ * KB * 2);              // 48 MB
    float*  C  = (float*)(ws + (size_t)B_ * KB * 2 + (size_t)NCOL * KB * 2);
    float*  Hf = (float*)((char*)C  + (size_t)B_ * NCOL * 4);      // 3 MB after C
    float*  GW = (float*)((char*)Hf + (size_t)B_ * HH * 4);

    prep_batch<<<B_, 128, 0, stream>>>(x, cond, hW1, hb1, gW1, gb1, gW2, gb2,
                                       Z, Hf, GW);
    pack_w<<<dim3(IN_, NEXP), 256, 0, stream>>>(hW2, Wb);
    gemm_kernel<<<dim3(B_/64, NCOL/64), 128, 0, stream>>>(Z, Wb, x, base_wp,
                                                          base_wn, hb2, C);
    epilogue<<<B_, 128, 0, stream>>>(C, Hf, GW, hW2, base_bn, hb2, out);
}
